// GCNN_61615600828570
// MI455X (gfx1250) — matrix operand order repaired and checked by reading, not compiled, not run on hardware
//
#include <hip/hip_runtime.h>
#include <hip/hip_bf16.h>
#include <math.h>

// ---------------- problem constants ----------------
constexpr int B  = 4;
constexpr int L  = 1024;
constexpr int E  = 140;
constexpr int D  = 140;
constexpr int NE = 10;   // edge types
constexpr int NU = 4;    // unRare types
constexpr int NB = 2;    // blocks
constexpr int ND = NU + 1; // distinct param sets (5)
constexpr int DP = 160;  // D padded to 10 WMMA n-tiles of 16
constexpr int KT = L / 32;               // 32 k-tiles per GEMM
constexpr int STEPS = 2 * ND * KT;       // 320 pipeline steps (dir x d x kt)
constexpr int SLAB_HALVES = 32 * DP;     // one k-slab of P: 32 x 160 f16
constexpr int SLAB_BYTES  = SLAB_HALVES * 2;          // 10240 B
constexpr int SLAB_CHUNKS = SLAB_BYTES / 16;          // 640 b128 chunks

typedef _Float16 v16h __attribute__((ext_vector_type(16)));
typedef _Float16 v8h  __attribute__((ext_vector_type(8)));
typedef float    v8f  __attribute__((ext_vector_type(8)));

__device__ __forceinline__ float sigmoidf_(float x) {
    return 1.0f / (1.0f + __expf(-x));
}

// ------------------------------------------------------------------
// 1) Projection: P[b][dir][d][l][0..DP) (f16, zero-padded cols) and
//    gate projection gp[b][dir][d][l] (f32).
//    grid = (L, B, 2*ND), block = DP(=160) threads.
// ------------------------------------------------------------------
__global__ void proj_kernel(const float* __restrict__ x,
                            const float* __restrict__ W_in,  const float* __restrict__ b_in,
                            const float* __restrict__ W_out, const float* __restrict__ b_out,
                            const float* __restrict__ Wg_in, const float* __restrict__ bg_in,
                            const float* __restrict__ Wg_out,const float* __restrict__ bg_out,
                            _Float16* __restrict__ P, float* __restrict__ gp, int blk)
{
    const int l = blockIdx.x, b = blockIdx.y;
    const int z = blockIdx.z;
    const int dir = z / ND, d = z % ND;
    const float* W  = (dir == 0 ? W_in  : W_out)  + (size_t)(blk * ND + d) * E * D;
    const float* bb = (dir == 0 ? b_in  : b_out)  + (size_t)(blk * ND + d) * D;
    const float* Wg = (dir == 0 ? Wg_in : Wg_out) + (size_t)(blk * ND + d) * E;
    const float  bg = (dir == 0 ? bg_in : bg_out)[blk * ND + d];

    __shared__ float xs[E];
    const int t = threadIdx.x;
    if (t < E) xs[t] = x[((size_t)b * L + l) * E + t];
    __syncthreads();

    float acc = 0.0f;
    if (t < D) {
        #pragma unroll 4
        for (int e = 0; e < E; ++e) acc += xs[e] * W[(size_t)e * D + t];
        acc += bb[t];
    }
    const size_t pofs = ((((size_t)b * 2 + dir) * ND + d) * L + l) * DP + t;
    P[pofs] = (t < D) ? (_Float16)acc : (_Float16)0.0f;

    if (t == 0) {
        float ga = bg;
        for (int e = 0; e < E; ++e) ga += xs[e] * Wg[e];
        gp[(((size_t)b * 2 + dir) * ND + d) * L + l] = ga;
    }
}

// ------------------------------------------------------------------
// 2a) In-gates: g[b][0][n][l] = sigmoid( sum_m adj[b,n,l,m] * gp_in[m] )
//     grid = (L, NE, B), block = 256 (coalesced row reduce).
// ------------------------------------------------------------------
__global__ void gate_in_kernel(const int* __restrict__ adj,
                               const float* __restrict__ gp,
                               float* __restrict__ g)
{
    const int l = blockIdx.x, n = blockIdx.y, b = blockIdx.z;
    const int d = (n < NU) ? n : NU;
    const int*   arow = adj + (((size_t)b * NE + n) * L + l) * L;
    const float* gpv  = gp + (((size_t)b * 2 + 0) * ND + d) * L;

    float s = 0.0f;
    for (int m = threadIdx.x; m < L; m += 256)
        s += arow[m] ? gpv[m] : 0.0f;

    __shared__ float red[256];
    red[threadIdx.x] = s;
    __syncthreads();
    for (int o = 128; o > 0; o >>= 1) {
        if (threadIdx.x < o) red[threadIdx.x] += red[threadIdx.x + o];
        __syncthreads();
    }
    if (threadIdx.x == 0)
        g[(((size_t)b * 2 + 0) * NE + n) * L + l] = sigmoidf_(red[0]);
}

// ------------------------------------------------------------------
// 2b) Out-gates: g[b][1][n][l] = sigmoid( sum_m adj[b,n,m,l] * gp_out[m] )
//     grid = (L/256, NE, B), block = 256; streams columns coalesced.
// ------------------------------------------------------------------
__global__ void gate_out_kernel(const int* __restrict__ adj,
                                const float* __restrict__ gp,
                                float* __restrict__ g)
{
    const int l = blockIdx.x * 256 + threadIdx.x;
    const int n = blockIdx.y, b = blockIdx.z;
    const int d = (n < NU) ? n : NU;
    const float* gpv   = gp + (((size_t)b * 2 + 1) * ND + d) * L;
    const int*   abase = adj + ((size_t)b * NE + n) * L * L + l;

    float s = 0.0f;
    #pragma unroll 4
    for (int m = 0; m < L; ++m)
        s += abase[(size_t)m * L] ? gpv[m] : 0.0f;

    g[(((size_t)b * 2 + 1) * NE + n) * L + l] = sigmoidf_(s);
}

// ------------------------------------------------------------------
// 3a) Gated adjacency (in): Apack[b][0][d][l][m] = sum_{n:et=d} g*adj
//     grid = (L, ND, B), block = 256.
// ------------------------------------------------------------------
__global__ void pack_in_kernel(const int* __restrict__ adj,
                               const float* __restrict__ g,
                               _Float16* __restrict__ Apack)
{
    const int l = blockIdx.x, d = blockIdx.y, b = blockIdx.z;
    const int nlo = (d < NU) ? d : NU;
    const int nhi = (d < NU) ? d + 1 : NE;

    for (int mm = threadIdx.x; mm < L; mm += 256) {
        float s = 0.0f;
        for (int n = nlo; n < nhi; ++n) {
            const float gg = g[(((size_t)b * 2 + 0) * NE + n) * L + l];
            s += adj[(((size_t)b * NE + n) * L + l) * L + mm] ? gg : 0.0f;
        }
        Apack[((((size_t)b * 2 + 0) * ND + d) * L + l) * L + mm] = (_Float16)s;
    }
}

// ------------------------------------------------------------------
// 3b) Gated adjacency (out, transposed): Apack[b][1][d][l][m] =
//     sum_{n:et=d} g_out[n][l] * adj[b,n,m,l].  LDS 32x32 tile transpose.
//     grid = (L/32, L/32, B*ND), block = (32, 8).
// ------------------------------------------------------------------
__global__ void pack_out_kernel(const int* __restrict__ adj,
                                const float* __restrict__ g,
                                _Float16* __restrict__ Apack)
{
    const int m0 = blockIdx.x * 32;
    const int l0 = blockIdx.y * 32;
    const int d  = blockIdx.z % ND;
    const int b  = blockIdx.z / ND;
    const int nlo = (d < NU) ? d : NU;
    const int nhi = (d < NU) ? d + 1 : NE;

    __shared__ float T[32][33];
    const int tx = threadIdx.x;   // 0..31 : l-local on load, m-local on store
    const int ty = threadIdx.y;   // 0..7

    float accv[4] = {0.0f, 0.0f, 0.0f, 0.0f};
    for (int n = nlo; n < nhi; ++n) {
        const float gv = g[(((size_t)b * 2 + 1) * NE + n) * L + (l0 + tx)];
        const int* abase = adj + ((size_t)b * NE + n) * L * L;
        #pragma unroll
        for (int k = 0; k < 4; ++k) {
            const int i = ty + 8 * k;  // m-local
            accv[k] += abase[(size_t)(m0 + i) * L + (l0 + tx)] ? gv : 0.0f;
        }
    }
    #pragma unroll
    for (int k = 0; k < 4; ++k) T[ty + 8 * k][tx] = accv[k];
    __syncthreads();

    #pragma unroll
    for (int k = 0; k < 4; ++k) {
        const int j = ty + 8 * k;  // l-local
        Apack[((((size_t)b * 2 + 1) * ND + d) * L + (l0 + j)) * L + (m0 + tx)] =
            (_Float16)T[tx][j];
    }
}

// ------------------------------------------------------------------
// 4) Main WMMA GEMM with async-LDS double-buffered B pipeline.
//    acc[b][l][0..DP) = sum_{dir,d} A'[b,dir,d] @ P[b,dir,d]
//
//    Block = 256 threads (8 waves) = 4 m-tiles x 2 n-groups; grid (16, B).
//    The 320 k-slabs of P for one batch form a LINEAR byte stream, so the
//    async prefetch source is a single pointer bumped by SLAB_BYTES/step.
//    Buffer select is kt&1 (segments are 32 slabs, even), made compile-time
//    by unrolling the k-loop by 2.
//
//    IMPORTANT: the slab pointer is passed into every async asm as a dummy
//    operand so it ESCAPES — otherwise LLVM's escape analysis concludes the
//    "memory" clobber can't touch the non-escaping __shared__ array and
//    hoists all B-fragment ds_loads out of the loop (seen in round 3).
//
//    Fragment layouts per CDNA5 ISA 7.12.2:
//      A 16x32 f16: lane<16 -> row=lane, K {k0..k0+7, k0+16..k0+23};
//                   lane>=16 -> row=lane-16, K {k0+8.., k0+24..}.
//      B 32x16 f16: lane -> N col (lane&15), K same as A (table order).
// ------------------------------------------------------------------
__global__ void __launch_bounds__(256, 1)
wmma_kernel(const _Float16* __restrict__ Apack,
            const _Float16* __restrict__ P,
            float* __restrict__ acc)
{
    // the only LDS object in this kernel -> LDS offset 0
    __shared__ __attribute__((aligned(64))) _Float16 slab[2][SLAB_HALVES];
    _Float16* slabp = &slab[0][0];   // escapes into the async asm below

    const int wave = threadIdx.x >> 5;
    const int lane = threadIdx.x & 31;
    const int mtile = blockIdx.x * 4 + (wave >> 1);  // 0..63
    const int ng = wave & 1;                         // n-group of 5 tiles
    const int b  = blockIdx.y;
    const int m0 = mtile * 16;
    const int n0 = ng * 5 * 16;                      // 0 or 80

    const _Float16* Pbase =
        P + (size_t)b * 2 * ND * (size_t)L * DP;     // [dir][d][L][DP]

    // async-copy one 10KB slab (640 b128 chunks = 2 full rounds + half round)
    const unsigned t16 = (unsigned)threadIdx.x * 16u;
    auto issue_slab_from = [&](const char* src, unsigned bufbase) {
        asm volatile("global_load_async_to_lds_b128 %0, %1, off"
                     :: "v"(bufbase + t16), "v"(src + t16), "s"(slabp)
                     : "memory");
        asm volatile("global_load_async_to_lds_b128 %0, %1, off"
                     :: "v"(bufbase + t16 + 4096u), "v"(src + t16 + 4096), "s"(slabp)
                     : "memory");
        if (threadIdx.x < 128)
            asm volatile("global_load_async_to_lds_b128 %0, %1, off"
                         :: "v"(bufbase + t16 + 8192u), "v"(src + t16 + 8192), "s"(slabp)
                         : "memory");
    };

    v8f c[5];
    #pragma unroll
    for (int nt = 0; nt < 5; ++nt)
        #pragma unroll
        for (int r = 0; r < 8; ++r) c[nt][r] = 0.0f;

    const int arow = m0 + (lane & 15);
    const int koff = (lane >> 4) * 8;
    const _Float16* Abase =
        Apack + (size_t)b * 2 * ND * (size_t)L * L + (size_t)arow * L;

    // prologue: stage slab 0; prefetch pointer then walks linearly
    const char* psrc = (const char*)Pbase;
    const char* pend = psrc + (size_t)STEPS * SLAB_BYTES;
    issue_slab_from(psrc, 0);
    psrc += SLAB_BYTES;
    asm volatile("s_wait_asynccnt 0x0" :: "s"(slabp) : "memory");
    __syncthreads();

    for (int dir = 0; dir < 2; ++dir) {
        for (int d = 0; d < ND; ++d) {
            const _Float16* Aseg =
                Abase + ((size_t)dir * ND + d) * (size_t)L * L;

            #pragma unroll 2
            for (int kt = 0; kt < KT; ++kt) {
                const int sel = kt & 1;              // compile-time per unroll half
                if (psrc < pend)
                    issue_slab_from(psrc, (unsigned)((sel ^ 1) * SLAB_BYTES));
                psrc += SLAB_BYTES;

                // --- A fragment: direct global load of this wave's 16 rows ---
                const _Float16* Ak = Aseg + kt * 32;
                __builtin_prefetch(Ak + 32, 0, 3);   // next k-chunk, keep near
                v8h alo = *(const v8h*)(Ak + koff);
                v8h ahi = *(const v8h*)(Ak + 16 + koff);
                v16h a = __builtin_shufflevector(alo, ahi,
                        0, 1, 2, 3, 4, 5, 6, 7, 8, 9, 10, 11, 12, 13, 14, 15);

                // --- B fragments from the LDS slab (one f16 per k row, row stride DP) ---
                const _Float16* prow = &slab[sel][0] + (lane & 15) + n0;
                v16h b0, b1, b2, b3, b4;
                #pragma unroll
                for (int s = 0; s < 16; ++s) {
                    const int kk = (s < 8) ? (koff + s) : (16 + koff + (s - 8));
                    const _Float16* pk = prow + kk * DP;
                    b0[s] = pk[0];
                    b1[s] = pk[16];
                    b2[s] = pk[32];
                    b3[s] = pk[48];
                    b4[s] = pk[64];
                }

                c[0] = __builtin_amdgcn_wmma_f32_16x16x32_f16(false, a, false, b0,
                                                              (short)0, c[0], false, false);
                c[1] = __builtin_amdgcn_wmma_f32_16x16x32_f16(false, a, false, b1,
                                                              (short)0, c[1], false, false);
                c[2] = __builtin_amdgcn_wmma_f32_16x16x32_f16(false, a, false, b2,
                                                              (short)0, c[2], false, false);
                c[3] = __builtin_amdgcn_wmma_f32_16x16x32_f16(false, a, false, b3,
                                                              (short)0, c[3], false, false);
                c[4] = __builtin_amdgcn_wmma_f32_16x16x32_f16(false, a, false, b4,
                                                              (short)0, c[4], false, false);

                // drain this iteration's async fill + fence buffer reuse
                asm volatile("s_wait_asynccnt 0x0" :: "s"(slabp) : "memory");
                __syncthreads();
            }
        }
    }

    // C/D layout: VGPR r, lanes 0-15: M=r, N=lane; lanes 16-31: M=8+r, N=lane-16
    float* out = acc + ((size_t)b * L + m0) * DP + n0;
    const int ncol = lane & 15;
    const int rofs = (lane >> 4) * 8;
    #pragma unroll
    for (int nt = 0; nt < 5; ++nt)
        #pragma unroll
        for (int r = 0; r < 8; ++r)
            out[(size_t)(r + rofs) * DP + nt * 16 + ncol] = c[nt][r];
}

// ------------------------------------------------------------------
// 5) x = relu(acc[:, :, :E]) + x ; optionally mirror into d_out.
// ------------------------------------------------------------------
__global__ void finish_kernel(const float* __restrict__ acc,
                              float* __restrict__ x,
                              float* __restrict__ out)
{
    const size_t i = (size_t)blockIdx.x * 256 + threadIdx.x;
    if (i >= (size_t)B * L * E) return;
    const size_t e  = i % E;
    const size_t bl = i / E;
    const float a  = acc[bl * DP + e];
    const float nx = (a > 0.0f ? a : 0.0f) + x[i];
    x[i] = nx;
    if (out) out[i] = nx;
}

// ------------------------------------------------------------------
// host-side orchestration
// ------------------------------------------------------------------
static inline size_t align256(size_t v) { return (v + 255) & ~(size_t)255; }

extern "C" void kernel_launch(void* const* d_in, const int* in_sizes, int n_in,
                              void* d_out, int out_size, void* d_ws, size_t ws_size,
                              hipStream_t stream)
{
    (void)in_sizes; (void)n_in; (void)out_size; (void)ws_size;

    const float* seq    = (const float*)d_in[0];
    const int*   adj    = (const int*)  d_in[1];
    const float* W_in   = (const float*)d_in[2];
    const float* b_in   = (const float*)d_in[3];
    const float* W_out  = (const float*)d_in[4];
    const float* b_out  = (const float*)d_in[5];
    const float* Wg_in  = (const float*)d_in[6];
    const float* bg_in  = (const float*)d_in[7];
    const float* Wg_out = (const float*)d_in[8];
    const float* bg_out = (const float*)d_in[9];

    // workspace layout
    char* ws = (char*)d_ws;
    size_t off = 0;
    float* xbuf = (float*)(ws + off);
    off = align256(off + (size_t)B * L * E * sizeof(float));
    float* accbuf = (float*)(ws + off);
    off = align256(off + (size_t)B * L * DP * sizeof(float));
    _Float16* P = (_Float16*)(ws + off);
    off = align256(off + (size_t)B * 2 * ND * L * DP * sizeof(_Float16));
    float* gp = (float*)(ws + off);
    off = align256(off + (size_t)B * 2 * ND * L * sizeof(float));
    float* g = (float*)(ws + off);
    off = align256(off + (size_t)B * 2 * NE * L * sizeof(float));
    _Float16* Apack = (_Float16*)(ws + off);
    off = align256(off + (size_t)B * 2 * ND * L * L * sizeof(_Float16));

    hipMemcpyAsync(xbuf, seq, (size_t)B * L * E * sizeof(float),
                   hipMemcpyDeviceToDevice, stream);

    for (int blk = 0; blk < NB; ++blk) {
        proj_kernel<<<dim3(L, B, 2 * ND), DP, 0, stream>>>(
            xbuf, W_in, b_in, W_out, b_out, Wg_in, bg_in, Wg_out, bg_out,
            P, gp, blk);

        gate_in_kernel<<<dim3(L, NE, B), 256, 0, stream>>>(adj, gp, g);
        gate_out_kernel<<<dim3(L / 256, NE, B), 256, 0, stream>>>(adj, gp, g);

        pack_in_kernel<<<dim3(L, ND, B), 256, 0, stream>>>(adj, g, Apack);
        pack_out_kernel<<<dim3(L / 32, L / 32, B * ND), dim3(32, 8), 0, stream>>>(
            adj, g, Apack);

        wmma_kernel<<<dim3(16, B), 256, 0, stream>>>(Apack, P, accbuf);

        const size_t nelem = (size_t)B * L * E;
        finish_kernel<<<(unsigned)((nelem + 255) / 256), 256, 0, stream>>>(
            accbuf, xbuf, (blk == NB - 1) ? (float*)d_out : nullptr);
    }
}
